// mamba_fusion_74148315398618
// MI455X (gfx1250) — matrix operand order repaired and checked by reading, not compiled, not run on hardware
//
#include <hip/hip_runtime.h>
#include <hip/hip_bf16.h>

// ---------------------------------------------------------------------------
// Mamba block, fully fused per batch row on gfx1250 (wave32, WMMA bf16).
// One workgroup (256 thr = 8 waves) per batch element; all intermediates in
// LDS (~159 KB -> 2 WGs / WGP with 320KB LDS). GEMMs via
// v_wmma_f32_16x16x32_bf16; scan parallel over the 256 inner channels.
// ---------------------------------------------------------------------------

#define D_MODEL  128
#define D_INNER  256
#define DT_RANK  8
#define D_STATE  4
#define BATCH    4096
#define SEQ      45
#define SEQP     48   // padded to 3 M-tiles of 16

typedef __attribute__((ext_vector_type(16))) __bf16 v16bf;
typedef __attribute__((ext_vector_type(8)))  float  v8f;

// ---- LDS layout (byte offsets), total 162816 B ----------------------------
#define OFF_XBF   0        // 48*128 bf16 = 12288   (A operand, GEMM1)
#define OFF_XC    12288    // 48*256 f32  = 49152   (xc, conv'd in place -> u)
#define OFF_ZBF   61440    // 48*256 bf16 = 24576   (gate z)
#define OFF_XDBL  86016    // 48*16  f32  = 3072    (dt|B|C projections)
#define OFF_YBF   89088    // 48*256 bf16 = 24576   (A operand, GEMM2)
#define OFF_FEAT  113664   // 48*128 f32  = 24576   (feat, for rep reduction)
#define OFF_WX    138240   // 16*256 f32  = 16384
#define OFF_WDT   154624   // 256*8  f32  = 8192
#define SMEM_BYTES 162816

// ---- WMMA operand gathers (ISA 7.12.2 layouts, wave32) ---------------------
// A: 16x32 bf16 tile from row-major LDS [rows][ldk].
// lane<16: M=lane,   K = {0..7, 16..23}; lane>=16: M=lane-16, K = {8..15, 24..31}
__device__ __forceinline__ v16bf load_a16(const __bf16* s, int mt, int kt,
                                          int ldk, int lane) {
  const __bf16* row = s + (mt * 16 + (lane & 15)) * ldk + kt * 32;
  const int kb = (lane >> 4) << 3;   // 0 or 8
  v16bf a;
#pragma unroll
  for (int t = 0; t < 16; ++t)
    a[t] = row[((t & 8) ? 16 : 0) + kb + (t & 7)];
  return a;
}

// B: 32x16 bf16 tile, element (k,n) = w[n*K + k] (w row-major [N][K]).
// lane<16: N=lane, K = {0..7, 16..23}; lane>=16: N=lane-16, K = {8..15, 24..31}
__device__ __forceinline__ v16bf load_b16(const __bf16* w, int n0, int kt,
                                          int K, int lane) {
  const __bf16* p = w + (size_t)(n0 + (lane & 15)) * K + kt * 32 + ((lane >> 4) << 3);
  v16bf b;
#pragma unroll
  for (int t = 0; t < 16; ++t) b[t] = p[((t & 8) ? 16 : 0) + (t & 7)];
  return b;
}

__device__ __forceinline__ v8f wmma_bf16(v16bf a, v16bf b, v8f c) {
  return __builtin_amdgcn_wmma_f32_16x16x32_bf16(
      /*neg_a=*/false, a, /*neg_b=*/false, b,
      /*c_mod=*/(short)0, c, /*reuse_a=*/false, /*reuse_b=*/false);
}

__device__ __forceinline__ float silu_f(float v) {
  return v / (1.f + __expf(-v));
}

// ---- kernel 0: pack W_in (512x128) and W_out (128x256) to bf16 in d_ws ----
__global__ __launch_bounds__(256) void pack_weights(const float* __restrict__ w_in,
                                                    const float* __restrict__ w_out,
                                                    __bf16* __restrict__ ws) {
  int i = blockIdx.x * 256 + threadIdx.x;
  const int NIN = 512 * 128;
  if (i < NIN)                    ws[i] = (__bf16)w_in[i];
  else if (i < NIN + 128 * 256)   ws[i] = (__bf16)w_out[i - NIN];
}

// ---- kernel 1: fused mamba block, one block per batch row ------------------
__global__ __launch_bounds__(256)
void mamba_fused(const float* __restrict__ x, const int* __restrict__ pad_mask,
                 const float* __restrict__ conv_w, const float* __restrict__ conv_b,
                 const float* __restrict__ W_x, const float* __restrict__ W_dt,
                 const float* __restrict__ b_dt, const float* __restrict__ A_log,
                 const float* __restrict__ Dvec,
                 const __bf16* __restrict__ w_in_bf,
                 const __bf16* __restrict__ w_out_bf,
                 float* __restrict__ out) {
  extern __shared__ char smem[];
  __bf16* s_xbf  = (__bf16*)(smem + OFF_XBF);
  float*  s_xc   = (float*)(smem + OFF_XC);
  __bf16* s_zbf  = (__bf16*)(smem + OFF_ZBF);
  float*  s_xdbl = (float*)(smem + OFF_XDBL);
  __bf16* s_ybf  = (__bf16*)(smem + OFF_YBF);
  float*  s_feat = (float*)(smem + OFF_FEAT);
  float*  s_wx   = (float*)(smem + OFF_WX);
  float*  s_wdt  = (float*)(smem + OFF_WDT);

  const int tid  = threadIdx.x;
  const int lane = tid & 31;
  const int wave = tid >> 5;
  const int bid  = blockIdx.x;
  const size_t xbase = (size_t)bid * SEQ * D_MODEL;

  // ---- P0: stage x[b] (zero-padded rows 45..47) + small weights ----------
  for (int i = tid; i < SEQP * D_MODEL; i += 256) {
    int row = i >> 7;
    float v = (row < SEQ) ? x[xbase + i] : 0.f;
    s_xbf[i] = (__bf16)v;
  }
  for (int i = tid; i < 16 * D_INNER; i += 256) s_wx[i] = W_x[i];
  for (int i = tid; i < D_INNER * DT_RANK; i += 256) s_wdt[i] = W_dt[i];
  __syncthreads();

  // ---- P1: xz = x @ W_in^T  (M=48, N=512, K=128) -------------------------
  // wave w owns n-tiles {w, w+8, w+16, w+24}; B tiles cached in VGPRs.
#pragma unroll 1
  for (int nn = 0; nn < 4; ++nn) {
    const int nt = wave + nn * 8;
    const v16bf b0 = load_b16(w_in_bf, nt * 16, 0, D_MODEL, lane);
    const v16bf b1 = load_b16(w_in_bf, nt * 16, 1, D_MODEL, lane);
    const v16bf b2 = load_b16(w_in_bf, nt * 16, 2, D_MODEL, lane);
    const v16bf b3 = load_b16(w_in_bf, nt * 16, 3, D_MODEL, lane);
#pragma unroll
    for (int mt = 0; mt < 3; ++mt) {
      v8f acc = {};
      acc = wmma_bf16(load_a16(s_xbf, mt, 0, D_MODEL, lane), b0, acc);
      acc = wmma_bf16(load_a16(s_xbf, mt, 1, D_MODEL, lane), b1, acc);
      acc = wmma_bf16(load_a16(s_xbf, mt, 2, D_MODEL, lane), b2, acc);
      acc = wmma_bf16(load_a16(s_xbf, mt, 3, D_MODEL, lane), b3, acc);
      const int r0 = mt * 16 + ((lane >> 4) << 3);
      if (nt < 16) {             // xc half (f32, conv needs precision)
        const int col = nt * 16 + (lane & 15);
#pragma unroll
        for (int r = 0; r < 8; ++r) s_xc[(r0 + r) * D_INNER + col] = acc[r];
      } else {                   // z half (bf16 gate)
        const int col = (nt - 16) * 16 + (lane & 15);
#pragma unroll
        for (int r = 0; r < 8; ++r) s_zbf[(r0 + r) * D_INNER + col] = (__bf16)acc[r];
      }
    }
  }
  __syncthreads();

  // ---- P2: depthwise conv (D_CONV=2) + SiLU, in place, descending l ------
  {
    const int e = tid;
    const float w0 = conv_w[e * 2 + 0], w1 = conv_w[e * 2 + 1], cb = conv_b[e];
#pragma unroll 1
    for (int l = SEQ - 1; l >= 0; --l) {
      float prev = (l > 0) ? s_xc[(l - 1) * D_INNER + e] : 0.f;
      float cur  = s_xc[l * D_INNER + e];
      s_xc[l * D_INNER + e] = silu_f(cb + prev * w0 + cur * w1);  // u[l][e]
    }
    for (int l = SEQ; l < SEQP; ++l) s_ybf[l * D_INNER + e] = (__bf16)0.f;
  }
  __syncthreads();

  // ---- P3: x_dbl = u @ W_x^T  (720 dots of length 256, VALU float4) ------
  for (int o = tid; o < SEQ * 16; o += 256) {
    const int l = o >> 4, r = o & 15;
    const float4* u4 = (const float4*)(s_xc + l * D_INNER);
    const float4* w4 = (const float4*)(s_wx + r * D_INNER);
    float acc = 0.f;
#pragma unroll 8
    for (int q = 0; q < D_INNER / 4; ++q) {
      float4 a = u4[q], w = w4[q];
      acc += a.x * w.x + a.y * w.y + a.z * w.z + a.w * w.w;
    }
    s_xdbl[o] = acc;
  }
  __syncthreads();

  // ---- P4: delta + selective scan + gate, thread e owns channel e --------
  {
    const int e = tid;
    const float an0 = -__expf(A_log[e * 4 + 0]);
    const float an1 = -__expf(A_log[e * 4 + 1]);
    const float an2 = -__expf(A_log[e * 4 + 2]);
    const float an3 = -__expf(A_log[e * 4 + 3]);
    const float bdt = b_dt[e], De = Dvec[e];
    float wdt[DT_RANK];
#pragma unroll
    for (int r = 0; r < DT_RANK; ++r) wdt[r] = s_wdt[e * DT_RANK + r];
    float h0 = 0.f, h1 = 0.f, h2 = 0.f, h3 = 0.f;
#pragma unroll 1
    for (int l = 0; l < SEQ; ++l) {
      const float* xd = s_xdbl + l * 16;
      float acc = bdt;
#pragma unroll
      for (int r = 0; r < DT_RANK; ++r) acc += xd[r] * wdt[r];
      const float dlt = (acc > 20.f) ? acc : log1pf(__expf(acc));  // softplus
      const float u_  = s_xc[l * D_INNER + e];
      const float du  = dlt * u_;
      h0 = __expf(dlt * an0) * h0 + du * xd[8];
      h1 = __expf(dlt * an1) * h1 + du * xd[9];
      h2 = __expf(dlt * an2) * h2 + du * xd[10];
      h3 = __expf(dlt * an3) * h3 + du * xd[11];
      const float y = h0 * xd[12] + h1 * xd[13] + h2 * xd[14] + h3 * xd[15]
                    + u_ * De;
      const float zv = (float)s_zbf[l * D_INNER + e];
      s_ybf[l * D_INNER + e] = (__bf16)(y * silu_f(zv));
    }
  }
  __syncthreads();

  // ---- P5: feat = y @ W_out^T  (M=48, N=128, K=256); stream to HBM -------
  {
    const int nt = wave;  // 8 waves = 8 n-tiles
    v16bf Bt[8];
#pragma unroll
    for (int kt = 0; kt < 8; ++kt)
      Bt[kt] = load_b16(w_out_bf, nt * 16, kt, D_INNER, lane);
    float* feat_g = out + (size_t)BATCH * D_MODEL + (size_t)bid * SEQ * D_MODEL;
#pragma unroll
    for (int mt = 0; mt < 3; ++mt) {
      v8f acc = {};
#pragma unroll
      for (int kt = 0; kt < 8; ++kt)
        acc = wmma_bf16(load_a16(s_ybf, mt, kt, D_INNER, lane), Bt[kt], acc);
      const int col = nt * 16 + (lane & 15);
      const int r0  = mt * 16 + ((lane >> 4) << 3);
#pragma unroll
      for (int r = 0; r < 8; ++r) {
        const int row = r0 + r;
        s_feat[row * D_MODEL + col] = acc[r];
        if (row < SEQ) feat_g[(size_t)row * D_MODEL + col] = acc[r];
      }
    }
  }
  __syncthreads();

  // ---- P6: rep = masked mean over L --------------------------------------
  if (tid < D_MODEL) {
    const int* pm = pad_mask + bid * SEQ;
    float num = 0.f, den = 0.f;
#pragma unroll 1
    for (int l = 0; l < SEQ; ++l) {
      const float c = 1.f - (float)pm[l];
      num += c * s_feat[l * D_MODEL + tid];
      den += c;
    }
    out[(size_t)bid * D_MODEL + tid] = num / den;
  }
}

extern "C" void kernel_launch(void* const* d_in, const int* in_sizes, int n_in,
                              void* d_out, int out_size, void* d_ws, size_t ws_size,
                              hipStream_t stream) {
  (void)in_sizes; (void)n_in; (void)out_size; (void)ws_size;
  const float* x      = (const float*)d_in[0];
  const int*   pm     = (const int*)d_in[1];
  const float* W_in   = (const float*)d_in[2];
  const float* conv_w = (const float*)d_in[3];
  const float* conv_b = (const float*)d_in[4];
  const float* W_x    = (const float*)d_in[5];
  const float* W_dt   = (const float*)d_in[6];
  const float* b_dt   = (const float*)d_in[7];
  const float* A_log  = (const float*)d_in[8];
  const float* Dvec   = (const float*)d_in[9];
  const float* W_out  = (const float*)d_in[10];

  __bf16* wbf = (__bf16*)d_ws;                 // 512*128 + 128*256 bf16 = 192 KiB
  const int n_pack = 512 * 128 + 128 * 256;    // 98304
  pack_weights<<<(n_pack + 255) / 256, 256, 0, stream>>>(W_in, W_out, wbf);

  hipFuncSetAttribute(reinterpret_cast<const void*>(&mamba_fused),
                      hipFuncAttributeMaxDynamicSharedMemorySize, SMEM_BYTES);
  mamba_fused<<<BATCH, 256, SMEM_BYTES, stream>>>(
      x, pm, conv_w, conv_b, W_x, W_dt, b_dt, A_log, Dvec,
      wbf, wbf + 512 * 128, (float*)d_out);
}
